// Block_14499809591347
// MI455X (gfx1250) — matrix operand order repaired and checked by reading, not compiled, not run on hardware
//
#include <hip/hip_runtime.h>

// ---------------------------------------------------------------------------
// Transformer block for MI455X (gfx1250, wave32, WMMA).
// All GEMMs / attention matmuls run on v_wmma_f32_16x16x32_bf16 (fp32 accum).
// Tile staging uses GLOBAL_LOAD_ASYNC_TO_LDS_B128 (ASYNCcnt) when the
// toolchain exposes the builtin, else falls back to sync b128 copies.
// ---------------------------------------------------------------------------

typedef __bf16 bf16;
typedef __attribute__((ext_vector_type(16))) __bf16 v16bf;
typedef __attribute__((ext_vector_type(8)))  float  v8f;
typedef int v4i_vs __attribute__((vector_size(16)));  // matches builtin param

#define DEV __device__ __forceinline__

#if defined(__has_builtin)
#if __has_builtin(__builtin_amdgcn_global_load_async_to_lds_b128)
#define HAVE_ASYNC_LDS 1
#endif
#endif

// Generic->addrspace casts via integer round trip (AS3 ptr = low 32 bits of
// the flat shared-aperture address; AS1 keeps the full 64-bit address).
DEV void async_copy_b128(const bf16* g, bf16* l) {
#ifdef HAVE_ASYNC_LDS
  __attribute__((address_space(1))) v4i_vs* gp =
      (__attribute__((address_space(1))) v4i_vs*)(unsigned long long)(size_t)g;
  __attribute__((address_space(3))) v4i_vs* lp =
      (__attribute__((address_space(3))) v4i_vs*)(unsigned int)(size_t)l;
  __builtin_amdgcn_global_load_async_to_lds_b128(gp, lp, 0, 0);
#else
  *reinterpret_cast<uint4*>(l) = *reinterpret_cast<const uint4*>(g);
#endif
}

DEV void async_wait0() {
#ifdef HAVE_ASYNC_LDS
#if __has_builtin(__builtin_amdgcn_s_wait_asynccnt)
  __builtin_amdgcn_s_wait_asynccnt(0);
#else
  asm volatile("s_wait_asynccnt 0" ::: "memory");
#endif
#endif
}

DEV v8f wmma_bf16(v16bf a, v16bf b, v8f c) {
  // (neg_a, A, neg_b, B, c_mod, C, reuse_a, reuse_b)
  return __builtin_amdgcn_wmma_f32_16x16x32_bf16(false, a, false, b, (short)0, c,
                                                 false, false);
}

// --- WMMA fragment loaders (contiguous k = 16*(lane/16) + s, wave32) ---------------------

// A fragment: 16(M) x 32(K), row-major source with element stride ld.
// Per lane: one contiguous 16-half run -> vectorizes to 2x ds_load_b128.
DEV v16bf frag_a(const bf16* p, int ld) {
  const int lane = threadIdx.x & 31;
  const bf16* row = p + (lane & 15) * ld;
  const int h16 = (lane >> 4) * 16;
  v16bf r;
#pragma unroll
  for (int v = 0; v < 4; ++v) {
    r[2 * v]         = row[h16 + 2 * v];
    r[2 * v + 1]     = row[h16 + 2 * v + 1];
    r[8 + 2 * v]     = row[h16 + 8 + 2 * v];
    r[8 + 2 * v + 1] = row[h16 + 8 + 2 * v + 1];
  }
  return r;
}

// B fragment: 32(K) x 16(N) from a K-major source (stride ld). Strided reads.
DEV v16bf frag_b(const bf16* p, int ld) {
  const int lane = threadIdx.x & 31;
  const int n = lane & 15;
  const int kb = (lane >> 4) * 16;
  v16bf r;
#pragma unroll
  for (int v = 0; v < 8; ++v) {
    r[2 * v]     = p[(kb + 2 * v) * ld + n];
    r[2 * v + 1] = p[(kb + 2 * v + 1) * ld + n];
  }
  return r;
}

// B fragment from an N-major (transposed) source: B[k][n] = p[n*ld + k].
// Per lane: one contiguous 16-half run -> vectorizes to 2x ds_load_b128.
DEV v16bf frag_bt(const bf16* p, int ld) {
  const int lane = threadIdx.x & 31;
  const bf16* row = p + (lane & 15) * ld + (lane >> 4) * 16;
  v16bf r;
#pragma unroll
  for (int v = 0; v < 16; ++v) r[v] = row[v];
  return r;
}

// --- half-wave (16-lane) row reductions for softmax (C-layout rows) ---------
DEV float half_red_max(float x) {
#pragma unroll
  for (int o = 8; o >= 1; o >>= 1) x = fmaxf(x, __shfl_xor(x, o, 32));
  return x;
}
DEV float half_red_sum(float x) {
#pragma unroll
  for (int o = 8; o >= 1; o >>= 1) x += __shfl_xor(x, o, 32);
  return x;
}

// ---------------------------------------------------------------------------
// fp32 -> bf16 conversion (weights), grid-stride
// ---------------------------------------------------------------------------
__global__ void f32_to_bf16_kernel(const float* __restrict__ src,
                                   bf16* __restrict__ dst, int n) {
  int i = blockIdx.x * blockDim.x + threadIdx.x;
  const int stride = gridDim.x * blockDim.x;
  for (; i < n; i += stride) dst[i] = (bf16)src[i];
}

// ---------------------------------------------------------------------------
// LayerNorm: one 256-thread block per row of D=1024, bf16 output
// ---------------------------------------------------------------------------
__global__ void __launch_bounds__(256) ln_kernel(const float* __restrict__ x,
                                                 const float* __restrict__ g,
                                                 const float* __restrict__ bia,
                                                 bf16* __restrict__ out) {
  constexpr int D = 1024;
  __shared__ float sh_s[8], sh_q[8];
  const int row = blockIdx.x;
  const int tid = threadIdx.x;
  const float* xr = x + (size_t)row * D;
  float v[4], s = 0.f, q = 0.f;
#pragma unroll
  for (int i = 0; i < 4; ++i) {
    v[i] = xr[tid + i * 256];
    s += v[i];
    q += v[i] * v[i];
  }
#pragma unroll
  for (int o = 16; o >= 1; o >>= 1) {
    s += __shfl_xor(s, o, 32);
    q += __shfl_xor(q, o, 32);
  }
  if ((tid & 31) == 0) { sh_s[tid >> 5] = s; sh_q[tid >> 5] = q; }
  __syncthreads();
  if (tid == 0) {
    float ts = 0.f, tq = 0.f;
#pragma unroll
    for (int w = 0; w < 8; ++w) { ts += sh_s[w]; tq += sh_q[w]; }
    sh_s[0] = ts; sh_q[0] = tq;
  }
  __syncthreads();
  const float mean = sh_s[0] * (1.0f / D);
  const float var  = sh_q[0] * (1.0f / D) - mean * mean;
  const float rs   = rsqrtf(var + 1e-5f);
  bf16* orow = out + (size_t)row * D;
#pragma unroll
  for (int i = 0; i < 4; ++i) {
    const int c = tid + i * 256;
    orow[c] = (bf16)((v[i] - mean) * rs * g[c] + bia[c]);
  }
}

// ---------------------------------------------------------------------------
// Tiled WMMA GEMM: C[M,N] = A[M,K](bf16) * B[K,N](bf16) + bias, epilogues:
//   MODE 0: bf16 store    MODE 1: fp32 store + residual    MODE 2: relu->bf16
// Block = 256 threads (8 waves, 4x2), tile 128x64, K-step 32.
// A tile staged async (layout-preserving); B tile staged transposed so both
// operand fragment loads are contiguous b128 LDS reads.
// ---------------------------------------------------------------------------
template <int MODE>
DEV void store_tile(v8f av, int gm0, int gn, int Ndim,
                    const float* __restrict__ bias,
                    const float* __restrict__ resid, void* __restrict__ outp) {
  const float bv = bias[gn];
#pragma unroll
  for (int r = 0; r < 8; ++r) {
    const size_t off = (size_t)(gm0 + r) * Ndim + gn;
    const float cval = av[r] + bv;
    if (MODE == 0)      ((bf16*)outp)[off] = (bf16)cval;
    else if (MODE == 1) ((float*)outp)[off] = cval + resid[off];
    else                ((bf16*)outp)[off] = (bf16)fmaxf(cval, 0.f);
  }
}

template <int MODE>
__global__ void __launch_bounds__(256) gemm_bf16_kernel(
    const bf16* __restrict__ A, const bf16* __restrict__ Bm,
    const float* __restrict__ bias, const float* __restrict__ resid,
    void* __restrict__ outp, int Mdim, int Ndim, int Kdim) {
  __shared__ bf16 As[128][40];  // M x K, 32+8 pad (row = 80B, 16B aligned)
  __shared__ bf16 Bt[64][40];   // N x K (transposed), same padding
  const int tid = threadIdx.x;
  const int wave = tid >> 5, lane = tid & 31;
  const int wm = wave & 3, wn = wave >> 2;
  const int m0 = blockIdx.x * 128;
  const int n0 = blockIdx.y * 64;
  v8f acc00 = {}, acc01 = {}, acc10 = {}, acc11 = {};

  for (int k0 = 0; k0 < Kdim; k0 += 32) {
    // A tile 128x32: async memory->LDS, one b128 lane-portion x2
#pragma unroll
    for (int c = 0; c < 2; ++c) {
      const int idx = tid + c * 256;
      const int r = idx >> 2, col = (idx & 3) * 8;
      async_copy_b128(A + (size_t)(m0 + r) * Kdim + k0 + col, &As[r][col]);
    }
    {  // B tile 32x64 -> LDS transposed (sync: needs per-element scatter)
      const int r = tid >> 3, nb = (tid & 7) * 8;
      const uint4 raw =
          *reinterpret_cast<const uint4*>(Bm + (size_t)(k0 + r) * Ndim + n0 + nb);
      const bf16* t = reinterpret_cast<const bf16*>(&raw);
#pragma unroll
      for (int j = 0; j < 8; ++j) Bt[nb + j][r] = t[j];
    }
    if (k0 + 32 < Kdim)  // pull next A rows toward L2 (global_prefetch_b8)
      __builtin_prefetch(A + (size_t)(m0 + (tid >> 2)) * Kdim + k0 + 32, 0, 0);
    async_wait0();
    __syncthreads();

    const v16bf a0 = frag_a(&As[wm * 32][0], 40);
    const v16bf a1 = frag_a(&As[wm * 32 + 16][0], 40);
    const v16bf b0 = frag_bt(&Bt[wn * 32][0], 40);
    const v16bf b1 = frag_bt(&Bt[wn * 32 + 16][0], 40);
    acc00 = wmma_bf16(a0, b0, acc00);
    acc01 = wmma_bf16(a0, b1, acc01);
    acc10 = wmma_bf16(a1, b0, acc10);
    acc11 = wmma_bf16(a1, b1, acc11);
    __syncthreads();
  }

  const int nlo = lane & 15, mhi = (lane >> 4) * 8;
  const int gmb = m0 + wm * 32 + mhi;
  const int gnb = n0 + wn * 32 + nlo;
  store_tile<MODE>(acc00, gmb,      gnb,      Ndim, bias, resid, outp);
  store_tile<MODE>(acc01, gmb,      gnb + 16, Ndim, bias, resid, outp);
  store_tile<MODE>(acc10, gmb + 16, gnb,      Ndim, bias, resid, outp);
  store_tile<MODE>(acc11, gmb + 16, gnb + 16, Ndim, bias, resid, outp);
}

// ---------------------------------------------------------------------------
// Flash-attention (causal). qkv: bf16 [B,S,3D]; y: bf16 [B,S,D].
// grid = (S/128, H, B); block = 256 (8 waves); wave owns 16 queries.
// hd = 64. QK^T = 4 WMMAs / 32 keys, PV = 4 WMMAs / 32 keys.
// K/V chunks staged async memory->LDS.
// ---------------------------------------------------------------------------
__global__ void __launch_bounds__(256) attn_kernel(const bf16* __restrict__ qkv,
                                                   bf16* __restrict__ y) {
  constexpr int S = 2048, D3 = 3072;
  __shared__ bf16 Ks[64][64];       // 8 KiB, row-major [key][hd]
  __shared__ bf16 Vs[64][64];       // 8 KiB
  __shared__ bf16 Ps[8][16 * 32];   // per-wave P-transpose staging, 8 KiB
  const int tid = threadIdx.x;
  const int wave = tid >> 5, lane = tid & 31;
  const int nlo = lane & 15, mhi = (lane >> 4) * 8;
  const int b = blockIdx.z, hh = blockIdx.y;
  const int q0 = blockIdx.x * 128;
  const int qrow = q0 + wave * 16;

  const bf16* qbase = qkv + (size_t)b * S * D3 + hh * 64;
  const bf16* kbase = qbase + 1024;
  const bf16* vbase = qbase + 2048;

  // Q fragments for this wave: 16 x 64 split into two K=32 A-frags
  const v16bf qa0 = frag_a(qbase + (size_t)qrow * D3, D3);
  const v16bf qa1 = frag_a(qbase + (size_t)qrow * D3 + 32, D3);

  v8f o0 = {}, o1 = {}, o2 = {}, o3 = {};
  float rowm[8], rowl[8];
#pragma unroll
  for (int r = 0; r < 8; ++r) { rowm[r] = -1e30f; rowl[r] = 0.f; }

  const int kend = q0 + 128;  // causal bound for this query block (<= S)
  for (int kc = 0; kc < kend; kc += 64) {
#pragma unroll
    for (int c = 0; c < 2; ++c) {  // async K/V chunk staging (b128 portions)
      const int idx = tid + c * 256;
      const int r = idx >> 3, col = (idx & 7) * 8;
      async_copy_b128(kbase + (size_t)(kc + r) * D3 + col, &Ks[r][col]);
      async_copy_b128(vbase + (size_t)(kc + r) * D3 + col, &Vs[r][col]);
    }
    async_wait0();
    __syncthreads();

#pragma unroll
    for (int j = 0; j < 2; ++j) {
      const int k0 = kc + j * 32;
      if (k0 <= qrow + 15) {  // wave-uniform: EXEC stays all-ones for WMMA
        v8f s0 = {}, s1 = {};
        s0 = wmma_bf16(qa0, frag_bt(&Ks[j * 32][0], 64), s0);
        s0 = wmma_bf16(qa1, frag_bt(&Ks[j * 32][32], 64), s0);
        s1 = wmma_bf16(qa0, frag_bt(&Ks[j * 32 + 16][0], 64), s1);
        s1 = wmma_bf16(qa1, frag_bt(&Ks[j * 32 + 16][32], 64), s1);

        const int key0 = k0 + nlo, key1 = k0 + 16 + nlo;
#pragma unroll
        for (int r = 0; r < 8; ++r) {
          const int qq = qrow + mhi + r;
          const float a  = (key0 <= qq) ? s0[r] * 0.125f : -1e30f;
          const float c2 = (key1 <= qq) ? s1[r] * 0.125f : -1e30f;
          const float mx = half_red_max(fmaxf(a, c2));
          const float nm = fmaxf(rowm[r], mx);
          const float sc = __expf(rowm[r] - nm);
          const float e0 = __expf(a - nm);
          const float e1 = __expf(c2 - nm);
          rowl[r] = rowl[r] * sc + half_red_sum(e0 + e1);
          rowm[r] = nm;
          o0[r] *= sc; o1[r] *= sc; o2[r] *= sc; o3[r] *= sc;
          Ps[wave][(mhi + r) * 32 + nlo]      = (bf16)e0;   // C -> A transpose
          Ps[wave][(mhi + r) * 32 + 16 + nlo] = (bf16)e1;
        }
        const v16bf pa = frag_a(&Ps[wave][0], 32);
        o0 = wmma_bf16(pa, frag_b(&Vs[j * 32][0],  64), o0);
        o1 = wmma_bf16(pa, frag_b(&Vs[j * 32][16], 64), o1);
        o2 = wmma_bf16(pa, frag_b(&Vs[j * 32][32], 64), o2);
        o3 = wmma_bf16(pa, frag_b(&Vs[j * 32][48], 64), o3);
      }
    }
    __syncthreads();
  }

  bf16* yout = y + ((size_t)b * S + qrow + mhi) * 1024 + hh * 64 + nlo;
#pragma unroll
  for (int r = 0; r < 8; ++r) {
    const float inv = 1.0f / rowl[r];
    bf16* p = yout + (size_t)r * 1024;
    p[0]  = (bf16)(o0[r] * inv);
    p[16] = (bf16)(o1[r] * inv);
    p[32] = (bf16)(o2[r] * inv);
    p[48] = (bf16)(o3[r] * inv);
  }
}

// ---------------------------------------------------------------------------
extern "C" void kernel_launch(void* const* d_in, const int* in_sizes, int n_in,
                              void* d_out, int out_size, void* d_ws, size_t ws_size,
                              hipStream_t stream) {
  (void)in_sizes; (void)n_in; (void)out_size; (void)ws_size;
  const float* x      = (const float*)d_in[0];
  const float* ln1_g  = (const float*)d_in[1];
  const float* ln1_b  = (const float*)d_in[2];
  const float* ln2_g  = (const float*)d_in[3];
  const float* ln2_b  = (const float*)d_in[4];
  const float* W_qkv  = (const float*)d_in[5];
  const float* b_qkv  = (const float*)d_in[6];
  const float* W_proj = (const float*)d_in[7];
  const float* b_proj = (const float*)d_in[8];
  const float* W_fc   = (const float*)d_in[9];
  const float* b_fc   = (const float*)d_in[10];
  const float* W_out  = (const float*)d_in[11];
  const float* b_out  = (const float*)d_in[12];
  float* out = (float*)d_out;

  // Workspace layout (64 MiB total, buffers reused across phases)
  char* ws = (char*)d_ws;
  bf16* wqkv_bf  = (bf16*)(ws);                       //  6 MiB
  bf16* wproj_bf = (bf16*)(ws + (size_t)(6u  << 20)); //  2 MiB
  bf16* wfc_bf   = (bf16*)(ws + (size_t)(8u  << 20)); //  8 MiB
  bf16* wout_bf  = (bf16*)(ws + (size_t)(16u << 20)); //  8 MiB
  bf16* h_bf     = (bf16*)(ws + (size_t)(24u << 20)); //  8 MiB (ln1/ln2 out)
  bf16* qkv_bf   = (bf16*)(ws + (size_t)(32u << 20)); // 24 MiB
  bf16* y_bf     = (bf16*)(ws + (size_t)(56u << 20)); //  8 MiB
  bf16* m1_bf    = (bf16*)(ws + (size_t)(32u << 20)); // 32 MiB (reuses qkv+y)

  // 1. weights -> bf16
  f32_to_bf16_kernel<<<2048, 256, 0, stream>>>(W_qkv,  wqkv_bf,  1024 * 3072);
  f32_to_bf16_kernel<<<2048, 256, 0, stream>>>(W_proj, wproj_bf, 1024 * 1024);
  f32_to_bf16_kernel<<<2048, 256, 0, stream>>>(W_fc,   wfc_bf,   1024 * 4096);
  f32_to_bf16_kernel<<<2048, 256, 0, stream>>>(W_out,  wout_bf,  4096 * 1024);

  // 2. h = LN1(x)
  ln_kernel<<<4096, 256, 0, stream>>>(x, ln1_g, ln1_b, h_bf);

  // 3. qkv = h @ W_qkv + b_qkv      [4096 x 3072]
  gemm_bf16_kernel<0><<<dim3(32, 48), 256, 0, stream>>>(
      h_bf, wqkv_bf, b_qkv, nullptr, qkv_bf, 4096, 3072, 1024);

  // 4. y = causal_attention(q, k, v)
  attn_kernel<<<dim3(16, 16, 2), 256, 0, stream>>>(qkv_bf, y_bf);

  // 5. x1 = y @ W_proj + b_proj + x   (fp32, stored in d_out)
  gemm_bf16_kernel<1><<<dim3(32, 16), 256, 0, stream>>>(
      y_bf, wproj_bf, b_proj, x, out, 4096, 1024, 1024);

  // 6. h2 = LN2(x1)
  ln_kernel<<<4096, 256, 0, stream>>>(out, ln2_g, ln2_b, h_bf);

  // 7. m1 = relu(h2 @ W_fc + b_fc)    [4096 x 4096]
  gemm_bf16_kernel<2><<<dim3(32, 64), 256, 0, stream>>>(
      h_bf, wfc_bf, b_fc, nullptr, m1_bf, 4096, 4096, 1024);

  // 8. out = m1 @ W_out + b_out + x1  (in-place residual, same-thread R/W)
  gemm_bf16_kernel<1><<<dim3(32, 16), 256, 0, stream>>>(
      m1_bf, wout_bf, b_out, out, out, 4096, 1024, 4096);
}
